// HypformerAttention_16320875725062
// MI455X (gfx1250) — matrix operand order repaired and checked by reading, not compiled, not run on hardware
//
#include <hip/hip_runtime.h>
#include <stdint.h>

// ---------------------------------------------------------------------------
// Hypformer Lorentz attention for MI455X (gfx1250), bf16 WMMA pipeline.
// B=8, N=2048, INPUT_DIM=512, HEAD_DIM=512, D=513 (time coord + 512 spatial)
// Round 2: stage A-operands via GLOBAL_LOAD_ASYNC_TO_LDS_B128 (ASYNCcnt) and
// add global_prefetch for the streaming B operands.
// ---------------------------------------------------------------------------

typedef __bf16 bf16_t;
typedef bf16_t v16bf __attribute__((ext_vector_type(16)));
typedef bf16_t v8bf  __attribute__((ext_vector_type(8)));
typedef float  v8f   __attribute__((ext_vector_type(8)));

constexpr int NBATCH = 8;
constexpr int NSEQ   = 2048;
constexpr int DK     = 512;           // spatial head dim (= input dim)
constexpr int BN     = NBATCH * NSEQ; // 16384 total rows
#define EPS_L 1e-8f

__device__ __forceinline__ unsigned short f2bf(float f) {
  unsigned int u = __float_as_uint(f);
  u += 0x7FFFu + ((u >> 16) & 1u);   // round-to-nearest-even
  return (unsigned short)(u >> 16);
}
__device__ __forceinline__ float bf2f(unsigned short h) {
  return __uint_as_float(((unsigned int)h) << 16);
}
__device__ __forceinline__ v16bf cat8(v8bf a, v8bf b) {
  return __builtin_shufflevector(a, b, 0,1,2,3,4,5,6,7,8,9,10,11,12,13,14,15);
}
__device__ __forceinline__ v8f wmma_bf16(v16bf a, v16bf b, v8f c) {
  // D(16x16 f32) = A(16x32 bf16) * B(32x16 bf16) + C
  return __builtin_amdgcn_wmma_f32_16x16x32_bf16(false, a, false, b,
                                                 (short)0, c, false, false);
}

// Async DMA of 16 bytes global -> LDS (per-lane), tracked by ASYNCcnt.
// lds_off = wave-relative LDS byte offset (addr[31:0] of the generic pointer).
__device__ __forceinline__ void async_copy_b128(unsigned int lds_off,
                                                const void* gsrc) {
  asm volatile("global_load_async_to_lds_b128 %0, %1, off"
               :
               : "v"(lds_off), "v"((unsigned long long)(uintptr_t)gsrc)
               : "memory");
}
__device__ __forceinline__ void wait_async0() {
  asm volatile("s_wait_asynccnt 0" ::: "memory");
}
__device__ __forceinline__ unsigned int lds_off_of(const void* p) {
  return (unsigned int)(uintptr_t)p;   // LDS aperture: addr[31:0] = LDS offset
}

// ---------------------------------------------------------------------------
// K0: fold ones-column into bias, convert weights to bf16.
//   W' [t][o][i] = W_t[o][1+i]  (bf16),  b'[t][o] = b_t[o] + W_t[o][0]
// ---------------------------------------------------------------------------
__global__ __launch_bounds__(256) void prep_weights(
    const float* __restrict__ Wq, const float* __restrict__ bq,
    const float* __restrict__ Wk, const float* __restrict__ bk,
    const float* __restrict__ Wv, const float* __restrict__ bv,
    unsigned short* __restrict__ wb, float* __restrict__ bp) {
  int idx = blockIdx.x * 256 + threadIdx.x;   // 3*512*512 threads
  int t = idx >> 18;
  int o = (idx >> 9) & 511;
  int i = idx & 511;
  const float* W = (t == 0) ? Wq : ((t == 1) ? Wk : Wv);
  const float* b = (t == 0) ? bq : ((t == 1) ? bk : bv);
  wb[idx] = f2bf(W[o * 513 + 1 + i]);
  if (i == 0) bp[t * 512 + o] = b[o] + W[o * 513];
}

// ---------------------------------------------------------------------------
// K1: x (f32) -> bf16
// ---------------------------------------------------------------------------
__global__ __launch_bounds__(256) void xf32_to_bf16(
    const float* __restrict__ x, unsigned short* __restrict__ xb) {
  int i = (blockIdx.x * 256 + threadIdx.x) * 4;
  float4 f = *(const float4*)(x + i);
  uint2 pk;
  pk.x = (unsigned int)f2bf(f.x) | ((unsigned int)f2bf(f.y) << 16);
  pk.y = (unsigned int)f2bf(f.z) | ((unsigned int)f2bf(f.w) << 16);
  *(uint2*)(xb + i) = pk;
}

// ---------------------------------------------------------------------------
// K2: QKV projection.  xs = x @ W'^T + b'   (M=16384, N=512, K=512)
// Block = 16-row slab x all 512 cols; 8 waves x 64 cols (4 WMMA tiles each).
// Full row in one block -> Lorentz time coord xt = sqrt(sum(xs^2)+1) reduced
// via LDS.  Stores bf16 xs and f32 time component.
// ---------------------------------------------------------------------------
__global__ __launch_bounds__(256) void qkv_gemm(
    const unsigned short* __restrict__ xb,
    const unsigned short* __restrict__ wb,
    const float* __restrict__ bp,
    unsigned short* __restrict__ qkvs,
    float* __restrict__ tvec) {
  __shared__ float ss[16];
  const int t    = blockIdx.y;            // 0=q 1=k 2=v
  const int rowb = blockIdx.x * 16;
  const int wave = threadIdx.x >> 5;
  const int lane = threadIdx.x & 31;
  const int lr = lane & 15, lh = lane >> 4;
  if (threadIdx.x < 16) ss[threadIdx.x] = 0.0f;
  __syncthreads();

  const unsigned short* wm   = wb + (size_t)t * 512 * 512;
  const unsigned short* arow = xb + (size_t)(rowb + lr) * DK + lh * 16;
  const int colw = wave * 64;

  v8f z = {0.f,0.f,0.f,0.f,0.f,0.f,0.f,0.f};
  v8f acc[4] = {z, z, z, z};

  #pragma unroll 4
  for (int k = 0; k < DK; k += 32) {
    // A fragment: lane-half K split {0-15} / {16-31}
    v16bf a = cat8(*(const v8bf*)(arow + k), *(const v8bf*)(arow + k + 8));
    #pragma unroll
    for (int tt = 0; tt < 4; ++tt) {
      const unsigned short* bptr =
          wm + (size_t)(colw + tt * 16 + lr) * DK + k + lh * 16;
      v16bf bfrag = *(const v16bf*)bptr;   // 16 contiguous K per lane-half
      acc[tt] = wmma_bf16(a, bfrag, acc[tt]);
    }
  }

  // Epilogue: bias add, store bf16, accumulate row sum-of-squares.
  unsigned short* qout = qkvs + (size_t)t * BN * DK;
  float rss[8] = {0, 0, 0, 0, 0, 0, 0, 0};
  #pragma unroll
  for (int tt = 0; tt < 4; ++tt) {
    int col = colw + tt * 16 + lr;
    float bc = bp[t * 512 + col];
    #pragma unroll
    for (int r = 0; r < 8; ++r) {
      float v = acc[tt][r] + bc;
      int row = rowb + r + lh * 8;
      qout[(size_t)row * DK + col] = f2bf(v);
      rss[r] += v * v;
    }
  }
  #pragma unroll
  for (int r = 0; r < 8; ++r) atomicAdd(&ss[r + lh * 8], rss[r]);
  __syncthreads();
  if (threadIdx.x < 16)
    tvec[(size_t)t * BN + rowb + threadIdx.x] = sqrtf(ss[threadIdx.x] + 1.0f);
}

// ---------------------------------------------------------------------------
// K3: per-batch score GEMM  S = qs @ ks^T  (2048x2048, K=512) + epilogue
//   att = (2 + 2*(qs.ks - q0*k0))/scale + bias
// 64x64 block tile, 8 waves as 4x2 grid, 2 WMMA tiles/wave.
// A (q) chunk staged per K-step via GLOBAL_LOAD_ASYNC_TO_LDS_B128 and read
// back with ds loads (shared by wave pairs); B (k rows) loaded directly
// (contiguous per lane) with global_prefetch ahead of the stream.
// ---------------------------------------------------------------------------
__global__ __launch_bounds__(256) void score_gemm(
    const unsigned short* __restrict__ qs,
    const unsigned short* __restrict__ ks,
    const float* __restrict__ tvec,
    const float* __restrict__ scale,
    const float* __restrict__ biasp,
    float* __restrict__ Sbuf) {
  __shared__ unsigned short aT[64 * 32];   // 64 rows x 32 K, row-major, 4KB
  const int b    = blockIdx.y;
  const int mt   = blockIdx.x >> 5;
  const int nt   = blockIdx.x & 31;
  const int tid  = threadIdx.x;
  const int wave = tid >> 5;
  const int lane = tid & 31;
  const int lr = lane & 15, lh = lane >> 4;
  const int rowb = mt * 64 + (wave >> 1) * 16;
  const int colb = nt * 64 + (wave & 1) * 32;

  const unsigned short* qb = qs + (size_t)b * NSEQ * DK;
  const unsigned short* kb = ks + (size_t)b * NSEQ * DK;
  const unsigned short* b0row = kb + (size_t)(colb + lr) * DK + lh * 16;
  const unsigned short* b1row = kb + (size_t)(colb + 16 + lr) * DK + lh * 16;

  // async staging source/dest for this thread: one 16B chunk of 64x32 tile
  const int srow = tid >> 2;           // 0..63
  const int scol = (tid & 3) * 8;      // 0..24
  const unsigned short* asrc0 = qb + (size_t)(mt * 64 + srow) * DK + scol;
  const unsigned int    adst  = lds_off_of(&aT[srow * 32 + scol]);

  // A-fragment LDS read pointers for this lane
  const unsigned short* afrag = &aT[((wave >> 1) * 16 + lr) * 32 + lh * 16];

  v8f z = {0.f,0.f,0.f,0.f,0.f,0.f,0.f,0.f};
  v8f acc0 = z, acc1 = z;

  for (int k = 0; k < DK; k += 32) {
    __syncthreads();                       // LDS free from previous step
    async_copy_b128(adst, asrc0 + k);      // DMA 64x32 q-chunk into LDS
    if (k + 128 < DK) {                    // prefetch streaming k rows
      __builtin_prefetch((const void*)(b0row + k + 128), 0, 0);
      __builtin_prefetch((const void*)(b1row + k + 128), 0, 0);
    }
    wait_async0();
    __syncthreads();                       // all waves' chunks landed

    v16bf a  = cat8(*(const v8bf*)afrag, *(const v8bf*)(afrag + 8));
    v16bf f0 = *(const v16bf*)(b0row + k);
    v16bf f1 = *(const v16bf*)(b1row + k);
    acc0 = wmma_bf16(a, f0, acc0);
    acc1 = wmma_bf16(a, f1, acc1);
  }

  const float* q0 = tvec + (size_t)b * NSEQ;            // time comps
  const float* k0 = tvec + (size_t)BN + (size_t)b * NSEQ;
  float invs = 1.0f / scale[0];
  float bia  = biasp[0];
  float q0r[8];
  #pragma unroll
  for (int r = 0; r < 8; ++r) q0r[r] = q0[rowb + r + lh * 8];
  float k0c0 = k0[colb + lr];
  float k0c1 = k0[colb + 16 + lr];

  float* Sb = Sbuf + (size_t)b * NSEQ * NSEQ;
  #pragma unroll
  for (int r = 0; r < 8; ++r) {
    int row = rowb + r + lh * 8;
    Sb[(size_t)row * NSEQ + colb + lr] =
        (2.f + 2.f * (acc0[r] - q0r[r] * k0c0)) * invs + bia;
    Sb[(size_t)row * NSEQ + colb + 16 + lr] =
        (2.f + 2.f * (acc1[r] - q0r[r] * k0c1)) * invs + bia;
  }
}

// ---------------------------------------------------------------------------
// K4: row softmax with mask; overwrites each f32 row in place with bf16
// probabilities (row pitch stays 4096 ushorts = 2048 f32).
// ---------------------------------------------------------------------------
__global__ __launch_bounds__(256) void softmax_rows(
    float* __restrict__ Sbuf, const unsigned char* __restrict__ mask) {
  __shared__ float red[256];
  const int row = blockIdx.x;        // 0..16383
  const int b   = row >> 11;
  const int tid = threadIdx.x;
  float* Srow = Sbuf + (size_t)row * NSEQ;
  const unsigned char* mrow = mask + (size_t)b * NSEQ;

  float v[8];
  float mx = -INFINITY;
  #pragma unroll
  for (int j = 0; j < 8; ++j) {
    int m = j * 256 + tid;
    float s = Srow[m];
    v[j] = mrow[m] ? s : -INFINITY;
    mx = fmaxf(mx, v[j]);
  }
  red[tid] = mx; __syncthreads();
  for (int s = 128; s > 0; s >>= 1) {
    if (tid < s) red[tid] = fmaxf(red[tid], red[tid + s]);
    __syncthreads();
  }
  mx = red[0]; __syncthreads();

  float sum = 0.f;
  #pragma unroll
  for (int j = 0; j < 8; ++j) {
    float e = (mx == -INFINITY) ? 0.f : __expf(v[j] - mx);
    v[j] = e;
    sum += e;
  }
  red[tid] = sum; __syncthreads();
  for (int s = 128; s > 0; s >>= 1) {
    if (tid < s) red[tid] += red[tid + s];
    __syncthreads();
  }
  float tot = red[0];
  float inv = (tot > 0.f) ? 1.0f / tot : 0.f;
  // all reads of Srow completed before the barriers above -> safe to overwrite
  unsigned short* prow = (unsigned short*)Srow;
  #pragma unroll
  for (int j = 0; j < 8; ++j) prow[j * 256 + tid] = f2bf(v[j] * inv);
}

// ---------------------------------------------------------------------------
// K5: PV GEMM + Lorentz midpoint.  Block = 16 queries x all 513 out dims.
// P chunk (16x32) staged via async DMA into LDS (shared by all 8 waves);
// V tiles (32 keys x 512 dims) transposed into LDS so B-fragments are
// contiguous ds reads.  ave0 (time comp) accumulated scalar-wise; epilogue
// normalizes by sqrt(max(|ave0^2 - sum(ave_s^2)|, eps)).
// ---------------------------------------------------------------------------
__global__ __launch_bounds__(256) void pv_gemm(
    const float* __restrict__ Sbuf,            // holds bf16 P rows, pitch 4096
    const unsigned short* __restrict__ vs,
    const float* __restrict__ tvec,
    float* __restrict__ out) {
  __shared__ unsigned short vT[512 * 40];      // 40KB, pitch 40 -> 16B aligned
  __shared__ unsigned short aP[16 * 32];       // 1KB P chunk, row-major
  __shared__ float ave0s[16];
  __shared__ float sss[16];
  const int b    = blockIdx.y;
  const int rowb = blockIdx.x * 16;
  const int tid  = threadIdx.x;
  const int wave = tid >> 5, lane = tid & 31;
  const int lr = lane & 15, lh = lane >> 4;

  const float* v0 = tvec + (size_t)2 * BN + (size_t)b * NSEQ;
  const unsigned short* P =
      (const unsigned short*)Sbuf + ((size_t)b * NSEQ + rowb) * (NSEQ * 2);

  if (tid < 16) { ave0s[tid] = 0.f; sss[tid] = 0.f; }
  __syncthreads();

  { // time-component: ave0[r] = sum_m P[r][m] * v0[m]
    int r = tid & 15, seg = tid >> 4;
    const unsigned short* pr = P + (size_t)r * (NSEQ * 2);
    float s = 0.f;
    for (int m = seg * 128; m < seg * 128 + 128; ++m)
      s += bf2f(pr[m]) * v0[m];
    atomicAdd(&ave0s[r], s);
  }

  const unsigned short* vb = vs + (size_t)b * NSEQ * DK;

  // async P staging: threads 0..63 each move one 16B chunk of the 16x32 tile
  const int prow_s = tid >> 2;          // 0..63 -> row 0..15 (tid<64)
  const int pcol_s = (tid & 3) * 8;
  const unsigned short* psrc = P + (size_t)prow_s * (NSEQ * 2) + pcol_s;
  const unsigned int    pdst = lds_off_of(&aP[prow_s * 32 + pcol_s]);

  // A-fragment LDS read pointer
  const unsigned short* afrag = &aP[lr * 32 + lh * 16];

  v8f z = {0.f,0.f,0.f,0.f,0.f,0.f,0.f,0.f};
  v8f acc[4] = {z, z, z, z};

  for (int k = 0; k < NSEQ; k += 32) {
    __syncthreads();
    if (tid < 64) async_copy_b128(pdst, psrc + k);   // DMA P chunk -> LDS
    // stage V tile (32 x 512) transposed into LDS
    #pragma unroll
    for (int i = 0; i < 8; ++i) {
      int c  = tid + i * 256;      // chunk 0..2047
      int m  = c >> 6;             // key row 0..31
      int cb = (c & 63) * 8;       // col base 0..504
      const unsigned short* src = vb + (size_t)(k + m) * DK + cb;
      uint4 raw = *(const uint4*)src;       // 8 contiguous bf16
      unsigned short e[8];
      e[0] = (unsigned short)raw.x; e[1] = (unsigned short)(raw.x >> 16);
      e[2] = (unsigned short)raw.y; e[3] = (unsigned short)(raw.y >> 16);
      e[4] = (unsigned short)raw.z; e[5] = (unsigned short)(raw.z >> 16);
      e[6] = (unsigned short)raw.w; e[7] = (unsigned short)(raw.w >> 16);
      #pragma unroll
      for (int j = 0; j < 8; ++j) vT[(cb + j) * 40 + m] = e[j];
    }
    wait_async0();
    __syncthreads();

    v16bf a = cat8(*(const v8bf*)afrag, *(const v8bf*)(afrag + 8));
    #pragma unroll
    for (int tt = 0; tt < 4; ++tt) {
      int d = wave * 64 + tt * 16 + lr;
      const unsigned short* l = &vT[d * 40 + lh * 16];
      v16bf bfrag = cat8(*(const v8bf*)l, *(const v8bf*)(l + 8));
      acc[tt] = wmma_bf16(a, bfrag, acc[tt]);
    }
  }

  // spatial sum-of-squares per row
  float rss[8] = {0, 0, 0, 0, 0, 0, 0, 0};
  #pragma unroll
  for (int tt = 0; tt < 4; ++tt)
    #pragma unroll
    for (int r = 0; r < 8; ++r) rss[r] += acc[tt][r] * acc[tt][r];
  #pragma unroll
  for (int r = 0; r < 8; ++r) atomicAdd(&sss[r + lh * 8], rss[r]);
  __syncthreads();

  // Lorentz midpoint normalization + store
  #pragma unroll
  for (int r = 0; r < 8; ++r) {
    int lrow = r + lh * 8;
    float a0 = ave0s[lrow], sq = sss[lrow];
    float den = sqrtf(fmaxf(fabsf(a0 * a0 - sq), EPS_L));
    float inv = 1.0f / den;
    size_t ob = ((size_t)b * NSEQ + rowb + lrow) * 513;
    #pragma unroll
    for (int tt = 0; tt < 4; ++tt) {
      int d = wave * 64 + tt * 16 + lr;
      out[ob + 1 + d] = acc[tt][r] * inv;
    }
  }
  if (tid < 16) {
    float a0 = ave0s[tid], sq = sss[tid];
    float den = sqrtf(fmaxf(fabsf(a0 * a0 - sq), EPS_L));
    out[((size_t)b * NSEQ + rowb + tid) * 513] = a0 / den;
  }
}

// ---------------------------------------------------------------------------
extern "C" void kernel_launch(void* const* d_in, const int* in_sizes, int n_in,
                              void* d_out, int out_size, void* d_ws,
                              size_t ws_size, hipStream_t stream) {
  const float* x            = (const float*)d_in[0];
  const unsigned char* mask = (const unsigned char*)d_in[1];
  const float* Wq = (const float*)d_in[2];
  const float* bq = (const float*)d_in[3];
  const float* Wk = (const float*)d_in[4];
  const float* bk = (const float*)d_in[5];
  const float* Wv = (const float*)d_in[6];
  const float* bv = (const float*)d_in[7];
  const float* scale = (const float*)d_in[8];
  const float* bias  = (const float*)d_in[9];
  float* out = (float*)d_out;

  char* ws = (char*)d_ws;
  size_t off = 0;
  auto alloc = [&](size_t bytes) -> char* {
    char* p = ws + off;
    off += (bytes + 255) & ~(size_t)255;
    return p;
  };
  unsigned short* wb   = (unsigned short*)alloc((size_t)3 * 512 * 512 * 2);
  float*          bp   = (float*)alloc((size_t)3 * 512 * 4);
  unsigned short* xb   = (unsigned short*)alloc((size_t)BN * DK * 2);
  unsigned short* qkvs = (unsigned short*)alloc((size_t)3 * BN * DK * 2);
  float*          tvec = (float*)alloc((size_t)3 * BN * 4);
  float*          Sbuf = (float*)alloc((size_t)NBATCH * NSEQ * NSEQ * 4);

  prep_weights<<<3072, 256, 0, stream>>>(Wq, bq, Wk, bk, Wv, bv, wb, bp);
  xf32_to_bf16<<<(BN * DK) / 1024, 256, 0, stream>>>(x, xb);
  qkv_gemm<<<dim3(BN / 16, 3), 256, 0, stream>>>(xb, wb, bp, qkvs, tvec);
  score_gemm<<<dim3(1024, NBATCH), 256, 0, stream>>>(
      qkvs, qkvs + (size_t)BN * DK, tvec, scale, bias, Sbuf);
  softmax_rows<<<BN, 256, 0, stream>>>(Sbuf, mask);
  pv_gemm<<<dim3(NSEQ / 16, NBATCH), 256, 0, stream>>>(
      Sbuf, qkvs + (size_t)2 * BN * DK, tvec, out);
}
